// our_RNN_22213570855246
// MI455X (gfx1250) — compile-verified, matrix operand order checked by reading, not run on hardware
//
#include <hip/hip_runtime.h>
#include <hip/hip_bf16.h>
#include <math.h>

// ---------------------------------------------------------------------------
// CDNA5 (gfx1250) wave32 WMMA RNN.
//   B=64, T=512, D_IN=256, H=512, D_OUT=128
// ---------------------------------------------------------------------------

typedef __attribute__((ext_vector_type(16))) __bf16 v16bf;
typedef __attribute__((ext_vector_type(8)))  float  v8f;
typedef __attribute__((ext_vector_type(4)))  unsigned int v4u;
typedef __attribute__((ext_vector_type(4)))  int v4i;

// pointer types matching the async-DMA builtin signatures
// (hipcc prints addrspace(1) as "__device__"; params are v4i pointers)
typedef __attribute__((address_space(1))) v4i g_v4i;   // global
typedef __attribute__((address_space(3))) v4i l_v4i;   // LDS

union FragBF { v16bf v; v4u q[2]; };

// ---- CDNA5 async global<->LDS DMA (ASYNCcnt path), with guarded fallbacks --
#if __has_builtin(__builtin_amdgcn_global_load_async_to_lds_b128)
#define HAVE_ASYNC_LD 1
#else
#define HAVE_ASYNC_LD 0
#endif
#if __has_builtin(__builtin_amdgcn_global_store_async_from_lds_b128)
#define HAVE_ASYNC_ST 1
#else
#define HAVE_ASYNC_ST 0
#endif

// Copy 32B (16 bf16) global -> LDS.
__device__ __forceinline__ void stage_ld32(__bf16* lds_dst, const __bf16* gsrc) {
#if HAVE_ASYNC_LD
  __builtin_amdgcn_global_load_async_to_lds_b128(
      (g_v4i*)gsrc, (l_v4i*)lds_dst, 0, 0);
  __builtin_amdgcn_global_load_async_to_lds_b128(
      (g_v4i*)(gsrc + 8), (l_v4i*)(lds_dst + 8), 0, 0);
#else
  ((v4u*)lds_dst)[0] = ((const v4u*)gsrc)[0];
  ((v4u*)lds_dst)[1] = ((const v4u*)gsrc)[1];
#endif
}

// Copy 32B (16 bf16) LDS -> global.
__device__ __forceinline__ void stage_st32(__bf16* gdst, const __bf16* lds_src) {
#if HAVE_ASYNC_ST
  __builtin_amdgcn_global_store_async_from_lds_b128(
      (g_v4i*)gdst, (l_v4i*)lds_src, 0, 0);
  __builtin_amdgcn_global_store_async_from_lds_b128(
      (g_v4i*)(gdst + 8), (l_v4i*)(lds_src + 8), 0, 0);
#else
  ((v4u*)gdst)[0] = ((const v4u*)lds_src)[0];
  ((v4u*)gdst)[1] = ((const v4u*)lds_src)[1];
#endif
}

__device__ __forceinline__ void wait_asynccnt0() {
#if HAVE_ASYNC_LD || HAVE_ASYNC_ST
#if __has_builtin(__builtin_amdgcn_s_wait_asynccnt)
  __builtin_amdgcn_s_wait_asynccnt(0);
#else
  asm volatile("s_wait_asynccnt 0x0" ::: "memory");
#endif
#endif
}

__device__ __forceinline__ void wait_dscnt0() {
#if __has_builtin(__builtin_amdgcn_s_wait_dscnt)
  __builtin_amdgcn_s_wait_dscnt(0);
#else
  asm volatile("s_wait_dscnt 0x0" ::: "memory");
#endif
}

__device__ __forceinline__ float fast_tanh(float x) {
#if __has_builtin(__builtin_amdgcn_tanhf)
  return __builtin_amdgcn_tanhf(x);            // native v_tanh_f32
#elif __has_builtin(__builtin_amdgcn_tanh_f32)
  return __builtin_amdgcn_tanh_f32(x);
#else
  return tanhf(x);
#endif
}

// A-fragment (16x32 bf16, MxK) per ISA table: lane = hi*16 + r holds row M=r,
// K = k0 + hi*8 .. +7 in VGPR0-3 and K = k0 + 16 + hi*8 .. +7 in VGPR4-7.
// B-fragment (32x16, KxN) layout == A-layout of B^T, so for C = A @ W^T both
// A (rows of A) and B (rows of W) load identically from row-major storage.
__device__ __forceinline__ v16bf load_frag_rowmajor(const __bf16* __restrict__ base,
                                                    int row0, int k0, int ld) {
  const int lane = threadIdx.x & 31;
  const int r    = lane & 15;
  const int hi   = lane >> 4;
  const __bf16* p = base + (size_t)(row0 + r) * (size_t)ld + (size_t)(k0 + hi * 8);
  FragBF f;
  f.q[0] = *(const v4u*)(p);        // global_load_b128
  f.q[1] = *(const v4u*)(p + 16);   // global_load_b128
  return f.v;
}

__device__ __forceinline__ v16bf load_frag_lds(const __bf16* p) {
  FragBF f;
  f.q[0] = *(const v4u*)(p);        // ds_load_b128
  f.q[1] = *(const v4u*)(p + 16);   // ds_load_b128
  return f.v;
}

__device__ __forceinline__ v8f wmma_bf16(v16bf a, v16bf b, v8f c) {
  // D(f32 16x16) = A(bf16 16x32) * B(bf16 32x16) + C
  return __builtin_amdgcn_wmma_f32_16x16x32_bf16(false, a, false, b,
                                                 (short)0, c, false, false);
}

// ---------------------------------------------------------------------------
// f32 -> bf16 elementwise convert
// ---------------------------------------------------------------------------
__global__ __launch_bounds__(256) void cvt_f32_bf16(const float* __restrict__ in,
                                                    __bf16* __restrict__ out, int n) {
  int i = blockIdx.x * 256 + threadIdx.x;
  if (i < n) out[i] = (__bf16)in[i];
}

// ---------------------------------------------------------------------------
// xproj = A(MxK) @ W(NxK)^T + bias0[n] + bias1[n], output bf16 (MxN).
// Block = 256 threads = 8 waves, each wave computes one 64x16 output strip.
// ---------------------------------------------------------------------------
__global__ __launch_bounds__(256) void gemm_xproj_bf16(
    const __bf16* __restrict__ A, const __bf16* __restrict__ W,
    const float* __restrict__ bias0, const float* __restrict__ bias1,
    __bf16* __restrict__ out, int M, int N, int K) {
  const int wave  = threadIdx.x >> 5;
  const int lane  = threadIdx.x & 31;
  const int nTile = N >> 4;
  const int strip = blockIdx.x * 8 + wave;
  const int m0 = (strip / nTile) * 64;
  const int n0 = (strip % nTile) * 16;
  if (m0 >= M) return;

  const v8f vzero = {0.f, 0.f, 0.f, 0.f, 0.f, 0.f, 0.f, 0.f};
  v8f acc[4] = {vzero, vzero, vzero, vzero};

  for (int k0 = 0; k0 < K; k0 += 32) {
    v16bf bf = load_frag_rowmajor(W, n0, k0, K);     // W rows n0..n0+15 (B = W^T)
#pragma unroll
    for (int mt = 0; mt < 4; ++mt) {
      v16bf af = load_frag_rowmajor(A, m0 + 16 * mt, k0, K);
      acc[mt] = wmma_bf16(af, bf, acc[mt]);
    }
  }

  const int r = lane & 15, hi = lane >> 4;
  const float bsum = bias0[n0 + r] + bias1[n0 + r];
#pragma unroll
  for (int mt = 0; mt < 4; ++mt)
#pragma unroll
    for (int v = 0; v < 8; ++v) {
      int m = m0 + 16 * mt + v + 8 * hi;             // C/D f32 VGPR layout
      out[(size_t)m * N + n0 + r] = (__bf16)(acc[mt][v] + bsum);
    }
}

// ---------------------------------------------------------------------------
// Recurrent scan: h_t = tanh(xproj_t + h_{t-1} @ Whh^T), H=512, B=64.
// ONE persistent workgroup (1024 thr = 32 waves on one WGP).
//  * hidden state double-buffered bf16 in LDS (2 x 64KB)
//  * xproj_t staged in LDS (2 x 64KB) via async global->LDS DMA, prefetched
//    one step ahead; each wave DMAs exactly the columns it consumes, so a
//    wave-local s_wait_asynccnt 0 is sufficient (no extra barriers).
//  * seq output drained via async LDS->global DMA of wave-private columns.
//  * per step per wave: 16 k-iters x (1 Whh frag from L2 + 4 h frags from
//    LDS + 4 WMMA) = 64 WMMAs; one workgroup barrier per step.
// ---------------------------------------------------------------------------
__global__ __launch_bounds__(1024) void rnn_scan_bf16(
    const __bf16* __restrict__ xproj,   // (B*T, H), row index = b*T + t
    const __bf16* __restrict__ Whh,     // (H, H) row-major
    __bf16* __restrict__ seq_out,       // (B, T, H) bf16 or nullptr
    __bf16* __restrict__ hlast,         // (B, H) bf16 or nullptr
    int T) {
  constexpr int Bd = 64, H = 512;
  __shared__ __bf16 hbuf[2][Bd][H];     // 2 x 64KB hidden state
  __shared__ __bf16 xpbuf[2][Bd][H];    // 2 x 64KB staged xproj_t

  const int wave = threadIdx.x >> 5;
  const int lane = threadIdx.x & 31;
  const int n0 = wave * 16;             // wave-private column range [n0, n0+16)
  const int r = lane & 15, hi = lane >> 4;
  const int bA = lane, bB = lane + 32;  // the two batch rows this lane stages

  // h_0 = 0
  {
    v4u z = {0u, 0u, 0u, 0u};
    v4u* dst = (v4u*)&hbuf[0][0][0];
    for (int i = threadIdx.x; i < (Bd * H * 2) / 16; i += 1024) dst[i] = z;
  }
  // stage xproj(t=0): wave w pulls its own columns for all 64 batch rows
  stage_ld32(&xpbuf[0][bA][n0], xproj + ((size_t)bA * T + 0) * H + n0);
  stage_ld32(&xpbuf[0][bB][n0], xproj + ((size_t)bB * T + 0) * H + n0);
  wait_asynccnt0();
  __syncthreads();

  for (int t = 0; t < T; ++t) {
    const int cur = t & 1, nxt = cur ^ 1;

    // async-prefetch next step's xproj while this step computes
    if (t + 1 < T) {
      stage_ld32(&xpbuf[nxt][bA][n0], xproj + ((size_t)bA * T + (t + 1)) * H + n0);
      stage_ld32(&xpbuf[nxt][bB][n0], xproj + ((size_t)bB * T + (t + 1)) * H + n0);
    }

    // init accumulators from staged input projection (LDS)
    v8f acc[4];
#pragma unroll
    for (int mt = 0; mt < 4; ++mt)
#pragma unroll
      for (int v = 0; v < 8; ++v) {
        int b = 16 * mt + v + 8 * hi;
        acc[mt][v] = (float)xpbuf[cur][b][n0 + r];
      }

    // h_{t-1} @ Whh^T : K = 512 in 16 WMMA steps
    for (int kt = 0; kt < 16; ++kt) {
      v16bf bf = load_frag_rowmajor(Whh, n0, kt * 32, H);   // L2-resident weights
#pragma unroll
      for (int mt = 0; mt < 4; ++mt) {
        v16bf af = load_frag_lds(&hbuf[cur][16 * mt + r][kt * 32 + hi * 8]);
        acc[mt] = wmma_bf16(af, bf, acc[mt]);
      }
    }

    // tanh (native v_tanh_f32) -> next hidden state in LDS
#pragma unroll
    for (int mt = 0; mt < 4; ++mt)
#pragma unroll
      for (int v = 0; v < 8; ++v) {
        int b = 16 * mt + v + 8 * hi;
        hbuf[nxt][b][n0 + r] = (__bf16)fast_tanh(acc[mt][v]);
      }

    // drain this step's hidden state (wave-private columns) to global
    if (seq_out) {
#if HAVE_ASYNC_ST
      wait_dscnt0();   // our LDS writes must land before the DMA engine reads
#endif
      stage_st32(seq_out + ((size_t)bA * T + t) * H + n0, &hbuf[nxt][bA][n0]);
      stage_st32(seq_out + ((size_t)bB * T + t) * H + n0, &hbuf[nxt][bB][n0]);
    }

    __syncthreads();     // publish hbuf[nxt] to all waves
    wait_asynccnt0();    // our xproj(t+1) chunks are now in LDS
  }

  if (hlast) {
    const v4u* src = (const v4u*)&hbuf[T & 1][0][0];
    v4u* dst = (v4u*)hlast;
    for (int i = threadIdx.x; i < (Bd * H * 2) / 16; i += 1024) dst[i] = src[i];
  }
}

// ---------------------------------------------------------------------------
// out(64x128 f32) = h_last(64x512 bf16) @ Wfc(128x512)^T + b_fc.
// One block, 8 waves; wave w -> N-tile w, 4 M-tiles each.
// ---------------------------------------------------------------------------
__global__ __launch_bounds__(256) void fc_bf16(
    const __bf16* __restrict__ hl, const __bf16* __restrict__ Wfc,
    const float* __restrict__ bias, float* __restrict__ out) {
  constexpr int K = 512, N = 128;
  const int wave = threadIdx.x >> 5;
  const int lane = threadIdx.x & 31;
  const int n0 = wave * 16;

  const v8f vzero = {0.f, 0.f, 0.f, 0.f, 0.f, 0.f, 0.f, 0.f};
  v8f acc[4] = {vzero, vzero, vzero, vzero};

  for (int k0 = 0; k0 < K; k0 += 32) {
    v16bf bf = load_frag_rowmajor(Wfc, n0, k0, K);
#pragma unroll
    for (int mt = 0; mt < 4; ++mt) {
      v16bf af = load_frag_rowmajor(hl, 16 * mt, k0, K);
      acc[mt] = wmma_bf16(af, bf, acc[mt]);
    }
  }

  const int r = lane & 15, hi = lane >> 4;
  const float bv = bias[n0 + r];
#pragma unroll
  for (int mt = 0; mt < 4; ++mt)
#pragma unroll
    for (int v = 0; v < 8; ++v) {
      int m = 16 * mt + v + 8 * hi;
      out[(size_t)m * N + n0 + r] = acc[mt][v] + bv;
    }
}

// ---------------------------------------------------------------------------
// Host launcher
// ---------------------------------------------------------------------------
extern "C" void kernel_launch(void* const* d_in, const int* in_sizes, int n_in,
                              void* d_out, int out_size, void* d_ws, size_t ws_size,
                              hipStream_t stream) {
  (void)in_sizes; (void)n_in; (void)out_size; (void)ws_size;
  constexpr int B = 64, T = 512, D_IN = 256, H = 512, D_OUT = 128;
  constexpr int M = B * T;

  const float* x     = (const float*)d_in[0];
  const float* w_ih0 = (const float*)d_in[1];
  const float* w_hh0 = (const float*)d_in[2];
  const float* b_ih0 = (const float*)d_in[3];
  const float* b_hh0 = (const float*)d_in[4];
  const float* w_ih1 = (const float*)d_in[5];
  const float* w_hh1 = (const float*)d_in[6];
  const float* b_ih1 = (const float*)d_in[7];
  const float* b_hh1 = (const float*)d_in[8];
  const float* w_fc  = (const float*)d_in[9];
  const float* b_fc  = (const float*)d_in[10];
  float* out = (float*)d_out;

  // workspace carve-up (all bf16 scratch), 256B aligned
  char* ws = (char*)d_ws;
  size_t off = 0;
  auto carve = [&](size_t bytes) -> char* {
    char* p = ws + off;
    off += (bytes + 255) & ~(size_t)255;
    return p;
  };
  __bf16* x_bf    = (__bf16*)carve((size_t)M * D_IN * 2);       // 16 MB
  __bf16* wih0_bf = (__bf16*)carve((size_t)H * D_IN * 2);
  __bf16* whh0_bf = (__bf16*)carve((size_t)H * H * 2);
  __bf16* wih1_bf = (__bf16*)carve((size_t)H * H * 2);
  __bf16* whh1_bf = (__bf16*)carve((size_t)H * H * 2);
  __bf16* wfc_bf  = (__bf16*)carve((size_t)D_OUT * H * 2);
  __bf16* xp0     = (__bf16*)carve((size_t)M * H * 2);          // 32 MB
  __bf16* h1seq   = (__bf16*)carve((size_t)M * H * 2);          // 32 MB
  __bf16* xp1     = (__bf16*)carve((size_t)M * H * 2);          // 32 MB
  __bf16* hlast   = (__bf16*)carve((size_t)B * H * 2);

  auto cvt = [&](const float* src, __bf16* dst, int n) {
    cvt_f32_bf16<<<(n + 255) / 256, 256, 0, stream>>>(src, dst, n);
  };
  cvt(x,     x_bf,    M * D_IN);
  cvt(w_ih0, wih0_bf, H * D_IN);
  cvt(w_hh0, whh0_bf, H * H);
  cvt(w_ih1, wih1_bf, H * H);
  cvt(w_hh1, whh1_bf, H * H);
  cvt(w_fc,  wfc_bf,  D_OUT * H);

  const int strips = (M / 64) * (H / 16);     // 16384
  const int gemmBlocks = strips / 8;          // 2048

  // layer 0: xproj0 = x @ w_ih0^T + b_ih0 + b_hh0
  gemm_xproj_bf16<<<gemmBlocks, 256, 0, stream>>>(x_bf, wih0_bf, b_ih0, b_hh0,
                                                  xp0, M, H, D_IN);
  // layer 0 scan -> full hidden sequence
  rnn_scan_bf16<<<1, 1024, 0, stream>>>(xp0, whh0_bf, h1seq, (__bf16*)nullptr, T);

  // layer 1: xproj1 = h1seq @ w_ih1^T + b_ih1 + b_hh1
  gemm_xproj_bf16<<<gemmBlocks, 256, 0, stream>>>(h1seq, wih1_bf, b_ih1, b_hh1,
                                                  xp1, M, H, H);
  // layer 1 scan -> only final hidden state
  rnn_scan_bf16<<<1, 1024, 0, stream>>>(xp1, whh1_bf, (__bf16*)nullptr, hlast, T);

  // final FC: out = hlast @ w_fc^T + b_fc
  fc_bf16<<<1, 256, 0, stream>>>(hlast, wfc_bf, b_fc, out);
}
